// MemoryModule_3994319585531
// MI455X (gfx1250) — matrix operand order repaired and checked by reading, not compiled, not run on hardware
//
#include <hip/hip_runtime.h>
#include <stdint.h>

// ---- shapes (fixed by the reference) ----
#define TT 4
#define BB 4
#define CK 128
#define CV 512
#define HH 64
#define WW 64
#define Q1 1024        // h*w
#define M1 4096        // T*h*w
#define PP 4096        // H*W

typedef _Float16 v4h  __attribute__((ext_vector_type(4)));
typedef _Float16 v8h  __attribute__((ext_vector_type(8)));
typedef _Float16 v16h __attribute__((ext_vector_type(16)));
typedef float    v8f  __attribute__((ext_vector_type(8)));

// ---------------- WMMA helpers ----------------
__device__ __forceinline__ v8f wmma_f16(v16h a, v16h b, v8f c) {
  // D = A(16x32 f16) * B(32x16 f16) + C(16x16 f32)
  return __builtin_amdgcn_wmma_f32_16x16x32_f16(false, a, false, b, (short)0, c,
                                                false, false);
}

__device__ __forceinline__ v16h cat16(v8h lo, v8h hi) {
  v16h r;
#pragma unroll
  for (int i = 0; i < 8; ++i) { r[i] = lo[i]; r[i + 8] = hi[i]; }
  return r;
}

__device__ __forceinline__ v8f zero8() {
  v8f z;
#pragma unroll
  for (int i = 0; i < 8; ++i) z[i] = 0.0f;
  return z;
}

// A fragment (16x32, f16) from an LDS tile stored [M rows][40 halves], K contiguous.
// Lane layout: row = lane%16; lanes<16 hold K 0..7 & 16..23, lanes>=16 hold K 8..15 & 24..31.
__device__ __forceinline__ v16h lds_a_frag(const _Float16* tile, int lane) {
  int row = lane & 15, k0 = (lane >> 4) * 8;
  const _Float16* p = tile + row * 40 + k0;
  return cat16(*(const v8h*)p, *(const v8h*)(p + 16));
}

// B fragment (32x16, f16) from an LDS tile stored [N rows][40 halves], K contiguous per N row.
// Lane layout: col = lane%16; lanes<16 hold K 0..7 & 16..23, lanes>=16 hold K 8..15 & 24..31.
__device__ __forceinline__ v16h lds_b_frag(const _Float16* tile, int lane) {
  int col = lane & 15, k0 = (lane >> 4) * 8;
  const _Float16* p = tile + col * 40 + k0;
  return cat16(*(const v8h*)p, *(const v8h*)(p + 16));
}

// A fragment directly from a global fp32 row (K contiguous), converting to f16.
__device__ __forceinline__ v16h glb_a_frag_f32(const float* row, int lane) {
  int k0 = (lane >> 4) * 8;
  const float4* p = (const float4*)(row + k0);
  float4 f0 = p[0], f1 = p[1];
  const float4* q = (const float4*)(row + k0 + 16);
  float4 f2 = q[0], f3 = q[1];
  v16h r;
  r[0] = (_Float16)f0.x;  r[1] = (_Float16)f0.y;  r[2] = (_Float16)f0.z;  r[3] = (_Float16)f0.w;
  r[4] = (_Float16)f1.x;  r[5] = (_Float16)f1.y;  r[6] = (_Float16)f1.z;  r[7] = (_Float16)f1.w;
  r[8] = (_Float16)f2.x;  r[9] = (_Float16)f2.y;  r[10] = (_Float16)f2.z; r[11] = (_Float16)f2.w;
  r[12] = (_Float16)f3.x; r[13] = (_Float16)f3.y; r[14] = (_Float16)f3.z; r[15] = (_Float16)f3.w;
  return r;
}

// A fragment directly from a global f16 row (K contiguous).
__device__ __forceinline__ v16h glb_a_frag_f16(const _Float16* row, int lane) {
  int k0 = (lane >> 4) * 8;
  return cat16(*(const v8h*)(row + k0), *(const v8h*)(row + k0 + 16));
}

// B fragment directly from a global f16 row for this lane's N column (K contiguous).
__device__ __forceinline__ v16h glb_b_frag_f16(const _Float16* rowk, int lane) {
  int k0 = (lane >> 4) * 8;
  return cat16(*(const v8h*)(rowk + k0), *(const v8h*)(rowk + k0 + 16));
}

// ---- vectorized LDS staging: 128 rows x 32 K elems -> [128][40] f16 tile ----
// fp32 source: float4 loads + packed cvt + b64 LDS stores. Uniform, fully unrolled.
__device__ __forceinline__ void stage128_f32(_Float16* dst, const float* src,
                                             size_t rowstride, int tid) {
#pragma unroll
  for (int it = 0; it < 4; ++it) {
    int i = tid + it * 256;             // 0..1023
    int rr = i >> 3, f0 = (i & 7) * 4;
    float4 f = *(const float4*)(src + (size_t)rr * rowstride + f0);
    v4h h;
    h[0] = (_Float16)f.x; h[1] = (_Float16)f.y;
    h[2] = (_Float16)f.z; h[3] = (_Float16)f.w;
    *(v4h*)(dst + rr * 40 + f0) = h;
  }
}

// f16 source: straight uint2 (b64) copies.
__device__ __forceinline__ void stage128_f16(_Float16* dst, const _Float16* src,
                                             size_t rowstride, int tid) {
#pragma unroll
  for (int it = 0; it < 4; ++it) {
    int i = tid + it * 256;             // 0..1023
    int rr = i >> 3, u2 = i & 7;
    const uint2* sp = (const uint2*)(src + (size_t)rr * rowstride);
    *(uint2*)(dst + rr * 40 + u2 * 4) = sp[u2];
  }
}

// =====================================================================
// K4: 2x2 average pool of query_value [B][Cv][64][64] -> qvpT [B][q=1024][v=512] f16
// (bilinear 64->32, align_corners=False is exactly 2x2 avg pooling; pooling commutes
//  with the later GEMM contraction over v, so we pool the B operand up front.)
// =====================================================================
__global__ __launch_bounds__(256) void k_pool(const float* __restrict__ qv,
                                              _Float16* __restrict__ qvpT) {
  int idx = blockIdx.x * 256 + threadIdx.x;        // B*Cv*32*32 = 2,097,152 threads
  int x = idx & 31, y = (idx >> 5) & 31, v = (idx >> 10) & 511, b = idx >> 19;
  const float* r0 = qv + ((((size_t)b * CV + v) * HH + 2 * y) * WW + 2 * x);
  float2 a = *(const float2*)r0;
  float2 d = *(const float2*)(r0 + WW);
  float s = 0.25f * (a.x + a.y + d.x + d.y);
  qvpT[((size_t)b * Q1 + y * 32 + x) * CV + v] = (_Float16)s;
}

// =====================================================================
// K1: part-1 scores  S[b][q][m] = sum_k qkl[b][k][q] * mkl[b][k][m]   (K=Ck=128)
// Workgroup tile: 64 q x 128 m. 8 waves: wave w -> q-strip (w>>1), m-half (w&1) (4 tiles).
// Both operands transposed through LDS (memory is [k][spatial] for both).
// =====================================================================
__global__ __launch_bounds__(256) void k_scores(const float* __restrict__ qkl,  // [B][Ck][1024]
                                                const float* __restrict__ mkl,  // [T][B][Ck][1024]
                                                _Float16* __restrict__ S) {
  int m0 = blockIdx.x * 128;
  int q0 = blockIdx.y * 64;
  int b  = blockIdx.z;
  int t  = m0 >> 10, s0 = m0 & 1023;
  __shared__ _Float16 As[64][40];    // [q][k-chunk]
  __shared__ _Float16 Bs[128][40];   // [m][k-chunk]
  int tid = threadIdx.x, lane = tid & 31, w = tid >> 5;
  int qs = w >> 1, mh = w & 1;
  v8f c[4];
#pragma unroll
  for (int j = 0; j < 4; ++j) c[j] = zero8();

  for (int kc = 0; kc < 4; ++kc) {
    int k0 = kc * 32;
    // stage A (transpose): 32k x 64q, float4 loads along q
#pragma unroll
    for (int it = 0; it < 2; ++it) {
      int i = tid + it * 256;          // 0..511
      int kk = i >> 4, q4 = (i & 15) * 4;
      float4 f = *(const float4*)(qkl + ((size_t)b * CK + k0 + kk) * Q1 + q0 + q4);
      As[q4 + 0][kk] = (_Float16)f.x;
      As[q4 + 1][kk] = (_Float16)f.y;
      As[q4 + 2][kk] = (_Float16)f.z;
      As[q4 + 3][kk] = (_Float16)f.w;
    }
    // stage B (transpose): 32k x 128m, float4 loads along m
#pragma unroll
    for (int it = 0; it < 4; ++it) {
      int i = tid + it * 256;          // 0..1023
      int kk = i >> 5, m4 = (i & 31) * 4;
      float4 f = *(const float4*)(mkl + (((size_t)t * BB + b) * CK + k0 + kk) * 1024 + s0 + m4);
      Bs[m4 + 0][kk] = (_Float16)f.x;
      Bs[m4 + 1][kk] = (_Float16)f.y;
      Bs[m4 + 2][kk] = (_Float16)f.z;
      Bs[m4 + 3][kk] = (_Float16)f.w;
    }
    __syncthreads();
    v16h a = lds_a_frag(&As[qs * 16][0], lane);
#pragma unroll
    for (int j = 0; j < 4; ++j)
      c[j] = wmma_f16(a, lds_b_frag(&Bs[(mh * 4 + j) * 16][0], lane), c[j]);
    __syncthreads();
  }
  int qbase = q0 + qs * 16 + ((lane >> 4) << 3);
#pragma unroll
  for (int j = 0; j < 4; ++j) {
    int m = m0 + (mh * 4 + j) * 16 + (lane & 15);
#pragma unroll
    for (int r = 0; r < 8; ++r)
      S[((size_t)b * Q1 + qbase + r) * M1 + m] = (_Float16)c[j][r];
  }
}

// =====================================================================
// K2: softmax over m (4096) for each (b,q) row of S, in place. fp32 math.
// Each thread owns 16 contiguous halves (2x b128 loads).
// =====================================================================
__global__ __launch_bounds__(256) void k_softmax1(_Float16* __restrict__ S) {
  int q = blockIdx.x, b = blockIdx.y;
  _Float16* row = S + ((size_t)b * Q1 + q) * M1;
  int tid = threadIdx.x;
  v8h* rp = (v8h*)(row + tid * 16);
  v8h h0 = rp[0], h1 = rp[1];
  float v[16];
  float mx = -1e30f;
#pragma unroll
  for (int i = 0; i < 8; ++i) {
    v[i] = (float)h0[i]; v[8 + i] = (float)h1[i];
  }
#pragma unroll
  for (int i = 0; i < 16; ++i) mx = fmaxf(mx, v[i]);
  __shared__ float red[256];
  red[tid] = mx; __syncthreads();
  for (int s = 128; s > 0; s >>= 1) {
    if (tid < s) red[tid] = fmaxf(red[tid], red[tid + s]);
    __syncthreads();
  }
  mx = red[0]; __syncthreads();
  float sum = 0.0f;
#pragma unroll
  for (int i = 0; i < 16; ++i) { v[i] = __expf(v[i] - mx); sum += v[i]; }
  red[tid] = sum; __syncthreads();
  for (int s = 128; s > 0; s >>= 1) {
    if (tid < s) red[tid] += red[tid + s];
    __syncthreads();
  }
  float inv = 1.0f / red[0];
#pragma unroll
  for (int i = 0; i < 8; ++i) {
    h0[i] = (_Float16)(v[i] * inv); h1[i] = (_Float16)(v[8 + i] * inv);
  }
  rp[0] = h0; rp[1] = h1;
}

// =====================================================================
// K3: memory[b][v][q] = sum_m mvl[b][v][m] * P1[b][q][m]    (K=m=4096)
// Tile 128v x 128q per WG. A fragments direct from global fp32 (m contiguous),
// B (f16 probabilities) double-buffered through LDS: one barrier per chunk.
// Writes d_out channels 512..1023.
// =====================================================================
__global__ __launch_bounds__(256) void k_out1(const float* __restrict__ mvl,    // [T][B][Cv][1024]
                                              const _Float16* __restrict__ P1,  // [B][1024][4096]
                                              float* __restrict__ out) {        // [B][1024][1024]
  int q0 = blockIdx.x * 128;
  int v0 = blockIdx.y * 128;
  int b  = blockIdx.z;
  __shared__ _Float16 Bs[2][128][40];   // [q][m-chunk], double buffered
  int tid = threadIdx.x, lane = tid & 31, w = tid >> 5;
  v8f c[8];
#pragma unroll
  for (int j = 0; j < 8; ++j) c[j] = zero8();
  int arow = v0 + w * 16 + (lane & 15);
  const _Float16* pbase = P1 + ((size_t)b * Q1 + q0) * M1;

  stage128_f16(&Bs[0][0][0], pbase, M1, tid);
  __syncthreads();
  for (int mc = 0; mc < 128; ++mc) {
    const _Float16* cur = &Bs[mc & 1][0][0];
    if (mc + 1 < 128)
      stage128_f16(&Bs[(mc + 1) & 1][0][0], pbase + (mc + 1) * 32, M1, tid);
    int m0c = mc * 32, t = m0c >> 10, ss = m0c & 1023;
    const float* ar = mvl + (((size_t)t * BB + b) * CV + arow) * 1024 + ss;
    v16h a = glb_a_frag_f32(ar, lane);
#pragma unroll
    for (int j = 0; j < 8; ++j)
      c[j] = wmma_f16(a, lds_b_frag(cur + j * 16 * 40, lane), c[j]);
    __syncthreads();
  }
  int ch0 = 512 + v0 + w * 16 + ((lane >> 4) << 3);
#pragma unroll
  for (int j = 0; j < 8; ++j) {
    int q = q0 + j * 16 + (lane & 15);
#pragma unroll
    for (int r = 0; r < 8; ++r)
      out[((size_t)b * 1024 + ch0 + r) * Q1 + q] = c[j][r];
  }
}

// =====================================================================
// K5: g_attn[b][t][k][v] = sum_p keys[t][b][k][p] * vals[t][b][v][p]  (K=p=4096)
// NT GEMM: both operands p-contiguous. A direct from global, B double-buffered LDS.
// Tile 128k x 128v per WG, one barrier per chunk.
// =====================================================================
__global__ __launch_bounds__(256) void k_gattn(const float* __restrict__ keys,  // [T][B][128][4096]
                                               const float* __restrict__ vals,  // [T][B][512][4096]
                                               float* __restrict__ G) {         // [B][T][128][512]
  int v0 = blockIdx.x * 128;
  int t  = blockIdx.y;
  int b  = blockIdx.z;
  __shared__ _Float16 Bs[2][128][40];   // [v][p-chunk], double buffered
  int tid = threadIdx.x, lane = tid & 31, w = tid >> 5;
  v8f c[8];
#pragma unroll
  for (int j = 0; j < 8; ++j) c[j] = zero8();
  const float* abase = keys + (((size_t)t * BB + b) * CK + w * 16 + (lane & 15)) * PP;
  const float* vbase = vals + (((size_t)t * BB + b) * CV + v0) * PP;

  stage128_f32(&Bs[0][0][0], vbase, PP, tid);
  __syncthreads();
  for (int pc = 0; pc < 128; ++pc) {
    const _Float16* cur = &Bs[pc & 1][0][0];
    if (pc + 1 < 128)
      stage128_f32(&Bs[(pc + 1) & 1][0][0], vbase + (pc + 1) * 32, PP, tid);
    v16h a = glb_a_frag_f32(abase + pc * 32, lane);
#pragma unroll
    for (int j = 0; j < 8; ++j)
      c[j] = wmma_f16(a, lds_b_frag(cur + j * 16 * 40, lane), c[j]);
    __syncthreads();
  }
  int kbase = w * 16 + ((lane >> 4) << 3);
#pragma unroll
  for (int j = 0; j < 8; ++j) {
    int v = v0 + j * 16 + (lane & 15);
#pragma unroll
    for (int r = 0; r < 8; ++r)
      G[(((size_t)b * TT + t) * CK + kbase + r) * CV + v] = c[j][r];
  }
}

// =====================================================================
// K6: softmax over T=4 of G -> f16 P2, same [b][t][k][v] layout.
// =====================================================================
__global__ __launch_bounds__(256) void k_softmaxT(const float* __restrict__ G,
                                                  _Float16* __restrict__ P2) {
  int idx = blockIdx.x * 256 + threadIdx.x;            // B*Ck*Cv = 262144 threads
  int v = idx & 511, k = (idx >> 9) & 127, b = idx >> 16;
  size_t base = (((size_t)b * TT) * CK + k) * CV + v;  // t-stride = CK*CV
  const size_t st = (size_t)CK * CV;
  float a0 = G[base], a1 = G[base + st], a2 = G[base + 2 * st], a3 = G[base + 3 * st];
  float mx = fmaxf(fmaxf(a0, a1), fmaxf(a2, a3));
  float e0 = __expf(a0 - mx), e1 = __expf(a1 - mx), e2 = __expf(a2 - mx), e3 = __expf(a3 - mx);
  float inv = 1.0f / (e0 + e1 + e2 + e3);
  P2[base]          = (_Float16)(e0 * inv);
  P2[base + st]     = (_Float16)(e1 * inv);
  P2[base + 2 * st] = (_Float16)(e2 * inv);
  P2[base + 3 * st] = (_Float16)(e3 * inv);
}

// =====================================================================
// K7: out[b][t*128+k][q] = sum_v P2[b][t][k][v] * qvpT[b][q][v]   (K=v=512)
// Pure-register GEMM: both operands are K-contiguous f16 in global -> direct
// fragment loads, no LDS, no barriers. Writes d_out channels 0..511.
// =====================================================================
__global__ __launch_bounds__(256) void k_out2(const _Float16* __restrict__ P2,    // [B][T][128][512]
                                              const _Float16* __restrict__ qvpT,  // [B][1024][512]
                                              float* __restrict__ out) {          // [B][1024][1024]
  int q0 = blockIdx.x * 128;
  int t  = blockIdx.y;
  int b  = blockIdx.z;
  int tid = threadIdx.x, lane = tid & 31, w = tid >> 5;
  v8f c[8];
#pragma unroll
  for (int j = 0; j < 8; ++j) c[j] = zero8();
  const _Float16* abase = P2 + (((size_t)b * TT + t) * CK + w * 16 + (lane & 15)) * CV;

#pragma unroll 4
  for (int vc = 0; vc < 16; ++vc) {
    int vv0 = vc * 32;
    v16h a = glb_a_frag_f16(abase + vv0, lane);
#pragma unroll
    for (int j = 0; j < 8; ++j) {
      const _Float16* brow = qvpT + ((size_t)b * Q1 + q0 + j * 16 + (lane & 15)) * CV + vv0;
      c[j] = wmma_f16(a, glb_b_frag_f16(brow, lane), c[j]);
    }
  }
  int ch0 = t * CK + w * 16 + ((lane >> 4) << 3);
#pragma unroll
  for (int j = 0; j < 8; ++j) {
    int q = q0 + j * 16 + (lane & 15);
#pragma unroll
    for (int r = 0; r < 8; ++r)
      out[((size_t)b * 1024 + ch0 + r) * Q1 + q] = c[j][r];
  }
}

// =====================================================================
extern "C" void kernel_launch(void* const* d_in, const int* in_sizes, int n_in,
                              void* d_out, int out_size, void* d_ws, size_t ws_size,
                              hipStream_t stream) {
  const float* memory_keys       = (const float*)d_in[0];  // [T,B,128,64,64]
  const float* memory_values     = (const float*)d_in[1];  // [T,B,512,64,64]
  const float* query_value       = (const float*)d_in[2];  // [B,512,64,64]
  const float* memory_keys_low   = (const float*)d_in[3];  // [T,B,128,32,32]
  const float* memory_values_low = (const float*)d_in[4];  // [T,B,512,32,32]
  const float* query_key_low     = (const float*)d_in[5];  // [B,128,32,32]
  float* out = (float*)d_out;                              // [B,1024,32,32]

  char* ws = (char*)d_ws;
  _Float16* S    = (_Float16*)(ws);                               // 32 MB: scores -> probs (in place)
  _Float16* qvpT = (_Float16*)(ws + ((size_t)32 << 20));          //  4 MB: pooled qv, [b][q][v]
  float*    G    = (float*)   (ws + ((size_t)36 << 20));          //  4 MB: g_attn fp32
  _Float16* P2   = (_Float16*)(ws + ((size_t)40 << 20));          //  2 MB: softmaxed g_attn f16

  k_pool    <<<dim3(8192),        256, 0, stream>>>(query_value, qvpT);
  k_scores  <<<dim3(32, 16, 4),   256, 0, stream>>>(query_key_low, memory_keys_low, S);
  k_softmax1<<<dim3(1024, 4),     256, 0, stream>>>(S);
  k_out1    <<<dim3(8, 4, 4),     256, 0, stream>>>(memory_values_low, S, out);
  k_gattn   <<<dim3(4, 4, 4),     256, 0, stream>>>(memory_keys, memory_values, G);
  k_softmaxT<<<dim3(1024),        256, 0, stream>>>(G, P2);
  k_out2    <<<dim3(8, 4, 4),     256, 0, stream>>>(P2, qvpT, out);
}
